// MultiHeadGATv2Dense_20246475834080
// MI455X (gfx1250) — matrix operand order repaired and checked by reading, not compiled, not run on hardware
//
#include <hip/hip_runtime.h>
#include <hip/hip_bf16.h>

typedef __attribute__((ext_vector_type(16))) __bf16 v16bf;
typedef __attribute__((ext_vector_type(8)))  float  v8f;
typedef __attribute__((ext_vector_type(2)))  float  v2f;
typedef __attribute__((ext_vector_type(4)))  unsigned int u32x4;
typedef __attribute__((ext_vector_type(8)))  int  i32x8;
typedef __attribute__((ext_vector_type(4)))  int  i32x4;

#define N_NODES  2048
#define IN_DIM   256
#define HEADS    6
#define HEAD_DIM 64
#define OUT_DIM  384
#define EPITCH   2052   // 2048 + 4 pad floats: row start bank rotates by 4

// LDS layout (floats)
#define OFF_E    0
#define OFF_QS   (16 * EPITCH)
#define OFF_A6   (OFF_QS + 16 * HEAD_DIM)
#define OFF_A4   (OFF_A6 + HEAD_DIM)
#define OFF_SCR  (OFF_A4 + HEAD_DIM)
#define LDS_FLOATS (OFF_SCR + 4096)

__device__ __forceinline__ unsigned f2bf1(float x) {
  union { float f; unsigned u; } v; v.f = x;
  unsigned r = v.u + 0x7FFFu + ((v.u >> 16) & 1u);
  return r >> 16;
}
__device__ __forceinline__ unsigned packbf(float lo, float hi) {
  return f2bf1(lo) | (f2bf1(hi) << 16);
}

union ABfrag { v16bf v; unsigned u[8]; };

// ---------------- prep: fp32 -> bf16 conversions -------------------------
__global__ void k_cvt_h(const float* __restrict__ h, unsigned short* __restrict__ hb) {
  int i = blockIdx.x * blockDim.x + threadIdx.x;
  if (i < N_NODES * IN_DIM) hb[i] = (unsigned short)f2bf1(h[i]);
}

__global__ void k_cvt_w(const float* __restrict__ Wsrc, const float* __restrict__ Wdst,
                        unsigned short* __restrict__ WsT, unsigned short* __restrict__ WdT) {
  int i = blockIdx.x * blockDim.x + threadIdx.x;
  if (i >= HEADS * IN_DIM * HEAD_DIM) return;
  int f  = i % HEAD_DIM;
  int d  = (i / HEAD_DIM) % IN_DIM;
  int hd = i / (HEAD_DIM * IN_DIM);
  int o  = (hd * HEAD_DIM + f) * IN_DIM + d;
  WsT[o] = (unsigned short)f2bf1(Wsrc[i]);
  WdT[o] = (unsigned short)f2bf1(Wdst[i]);
}

// ---------------- projection: Q = h*Wsrc, K = h*Wdst via WMMA ------------
__global__ void __launch_bounds__(128) k_proj(
    const unsigned short* __restrict__ hb,
    const unsigned short* __restrict__ WsT,
    const unsigned short* __restrict__ WdT,
    float* __restrict__ Qf, float* __restrict__ Kf,
    unsigned short* __restrict__ KT)
{
  const int i0   = blockIdx.x * 16;
  const int hd   = blockIdx.y;
  const int wave = threadIdx.x >> 5;
  const int lane = threadIdx.x & 31;
  const int half = lane >> 4;
  const int mrow = lane & 15;

  const int row  = i0 + mrow;
  const int fcol = wave * 16 + mrow;
  const unsigned* hrow = (const unsigned*)(hb + (size_t)row * IN_DIM);
  const unsigned* ws   = (const unsigned*)(WsT + ((size_t)hd * HEAD_DIM + fcol) * IN_DIM);
  const unsigned* wd   = (const unsigned*)(WdT + ((size_t)hd * HEAD_DIM + fcol) * IN_DIM);

  v8f cq = {}; v8f ck = {};
#pragma unroll
  for (int kc = 0; kc < IN_DIM / 32; ++kc) {
    ABfrag a, bs, bd;
    const uint4* pa = (const uint4*)(hrow + kc * 16 + half * 8);
    uint4 a0 = pa[0];
    uint4 a1 = pa[1];
    a.u[0] = a0.x; a.u[1] = a0.y; a.u[2] = a0.z; a.u[3] = a0.w;
    a.u[4] = a1.x; a.u[5] = a1.y; a.u[6] = a1.z; a.u[7] = a1.w;
    const uint4* ps = (const uint4*)(ws + kc * 16 + half * 8);
    uint4 s0 = ps[0], s1 = ps[1];
    bs.u[0] = s0.x; bs.u[1] = s0.y; bs.u[2] = s0.z; bs.u[3] = s0.w;
    bs.u[4] = s1.x; bs.u[5] = s1.y; bs.u[6] = s1.z; bs.u[7] = s1.w;
    const uint4* pd = (const uint4*)(wd + kc * 16 + half * 8);
    uint4 d0 = pd[0], d1 = pd[1];
    bd.u[0] = d0.x; bd.u[1] = d0.y; bd.u[2] = d0.z; bd.u[3] = d0.w;
    bd.u[4] = d1.x; bd.u[5] = d1.y; bd.u[6] = d1.z; bd.u[7] = d1.w;

    cq = __builtin_amdgcn_wmma_f32_16x16x32_bf16(false, a.v, false, bs.v, (short)0, cq, false, false);
    ck = __builtin_amdgcn_wmma_f32_16x16x32_bf16(false, a.v, false, bd.v, (short)0, ck, false, false);
  }

#pragma unroll
  for (int v = 0; v < 8; ++v) {
    int n = i0 + v + half * 8;
    int f = wave * 16 + mrow;
    size_t qo = ((size_t)hd * N_NODES + n) * HEAD_DIM + f;
    Qf[qo] = cq[v];
    Kf[qo] = ck[v];
    KT[((size_t)hd * HEAD_DIM + f) * N_NODES + n] = (unsigned short)f2bf1(ck[v]);
  }
}

// ---------------- fused attention: logits + softmax + P@K ----------------
__global__ void __launch_bounds__(256) k_attn(
    const float* __restrict__ Qf, const float* __restrict__ Kf,
    const unsigned short* __restrict__ KT,
    const float* __restrict__ adj, const float* __restrict__ avec,
    const float* __restrict__ bias,
    float* __restrict__ hprime, float* __restrict__ attn)
{
  extern __shared__ float smem[];
  float* lds_e = smem + OFF_E;
  float* Qs    = smem + OFF_QS;
  float* As6   = smem + OFF_A6;
  float* As4   = smem + OFF_A4;
  float* dscr  = smem + OFF_SCR;

  const int i0   = blockIdx.x * 16;
  const int hd   = blockIdx.y;
  const int tid  = threadIdx.x;
  const int wave = tid >> 5;
  const int lane = tid & 31;

  // --- TDM: DMA the 16x64 fp32 Q tile straight into LDS (wave 0 issues) ---
  if (wave == 0) {
    unsigned long long ga = (unsigned long long)(uintptr_t)(Qf)
                          + (((size_t)hd * N_NODES + i0) * HEAD_DIM) * sizeof(float);
    u32x4 g0;
    g0.x = 1u;                                   // count=1, user descriptor
    g0.y = (unsigned)(OFF_QS * sizeof(float));   // lds_addr (bytes)
    g0.z = (unsigned)(ga & 0xFFFFFFFFull);       // global_addr[31:0]
    g0.w = (unsigned)((ga >> 32) & 0x01FFFFFFull) | (2u << 30);  // addr[56:32] | type=2
    i32x8 g1;
    g1[0] = (int)(2u << 16);                     // data_size=2 (4B)
    g1[1] = (int)(((unsigned)HEAD_DIM & 0xFFFFu) << 16);   // tensor_dim0[15:0]=64
    g1[2] = (int)(((unsigned)N_NODES & 0xFFFFu) << 16);    // dim0 hi=0 | tensor_dim1 lo=2048
    g1[3] = (int)(((unsigned)HEAD_DIM) << 16);   // tensor_dim1 hi=0 | tile_dim0=64
    g1[4] = 16;                                  // tile_dim1=16 | tile_dim2=0
    g1[5] = HEAD_DIM;                            // tensor_dim0_stride = 64
    g1[6] = 0;
    g1[7] = 0;
    i32x4 gz4 = {0, 0, 0, 0};
    i32x8 gz8 = {0, 0, 0, 0, 0, 0, 0, 0};
    __builtin_amdgcn_tensor_load_to_lds(g0, g1, gz4, gz4, gz8, 0);
    __builtin_amdgcn_s_wait_tensorcnt(0);
  }
  if (tid < HEAD_DIM) {
    float av = avec[hd * HEAD_DIM + tid];
    As6[tid] = 0.6f * av;
    As4[tid] = 0.4f * av;
  }
  __syncthreads();

  // ---- phase A: e[r][j] = sum_f leaky(Q+K)*a, leaky(x)=0.6x+0.4|x| ----
  for (int jj = 0; jj < N_NODES / 256; ++jj) {
    const int j = jj * 256 + tid;
    const float* krow = Kf + ((size_t)hd * N_NODES + j) * HEAD_DIM;
    __builtin_prefetch(krow + 256 * HEAD_DIM, 0, 0);
    const v2f zero2 = {0.f, 0.f};
    v2f acc2[16];
#pragma unroll
    for (int r = 0; r < 16; ++r) acc2[r] = zero2;
#pragma unroll
    for (int fc = 0; fc < 4; ++fc) {
      v2f k2[8], a6[8], a4[8];
      const v2f* kp = (const v2f*)(krow + fc * 16);
      const v2f* a6p = (const v2f*)(As6 + fc * 16);
      const v2f* a4p = (const v2f*)(As4 + fc * 16);
#pragma unroll
      for (int t = 0; t < 8; ++t) { k2[t] = kp[t]; a6[t] = a6p[t]; a4[t] = a4p[t]; }
#pragma unroll
      for (int r = 0; r < 16; ++r) {
        const v2f* q2 = (const v2f*)(Qs + r * HEAD_DIM + fc * 16);
        v2f s = acc2[r];
#pragma unroll
        for (int t = 0; t < 8; ++t) {
          v2f x = q2[t] + k2[t];                      // v_pk_add_f32
          s = __builtin_elementwise_fma(a6[t], x, s); // v_pk_fma_f32
          s = __builtin_elementwise_fma(a4[t], __builtin_elementwise_abs(x), s); // abs = src mod
        }
        acc2[r] = s;
      }
    }
#pragma unroll
    for (int r = 0; r < 16; ++r) {
      float e = acc2[r].x + acc2[r].y;
      float m = adj[(size_t)(i0 + r) * N_NODES + j];
      lds_e[r * EPITCH + j] = (m > 0.f) ? e : -1e9f;
    }
  }
  __syncthreads();

  // ---- phase B: exact softmax per row (2 rows per wave) + attn_mean ----
  for (int r = wave * 2; r < wave * 2 + 2; ++r) {
    float m = -3.4e38f;
    for (int j = lane; j < N_NODES; j += 32) m = fmaxf(m, lds_e[r * EPITCH + j]);
#pragma unroll
    for (int off = 16; off; off >>= 1) m = fmaxf(m, __shfl_xor(m, off));
    float s = 0.f;
    for (int j = lane; j < N_NODES; j += 32) {
      float p = __expf(lds_e[r * EPITCH + j] - m);
      lds_e[r * EPITCH + j] = p;
      s += p;
    }
#pragma unroll
    for (int off = 16; off; off >>= 1) s += __shfl_xor(s, off);
    const float inv = __builtin_amdgcn_rcpf(s);
    for (int j = lane; j < N_NODES; j += 32) {
      float p = lds_e[r * EPITCH + j] * inv;
      lds_e[r * EPITCH + j] = p;
      atomicAdd(&attn[(size_t)(i0 + r) * N_NODES + j], p * (1.0f / HEADS));
    }
  }
  __syncthreads();

  // ---- phase D: h' = P(16x2048) @ K(2048x64), all 8 waves, split j ----
  {
    const int w4   = wave & 3;        // f-tile
    const int jpar = wave >> 2;       // j parity
    const int half = lane >> 4;
    const int mrow = lane & 15;
    const int fcol = w4 * 16 + mrow;
    const unsigned* ktp = (const unsigned*)(KT + ((size_t)hd * HEAD_DIM + fcol) * N_NODES);
    v8f c = {};
#pragma unroll 2
    for (int jt = jpar; jt < N_NODES / 32; jt += 2) {
      ABfrag a, b;
      const float* pe = lds_e + mrow * EPITCH + jt * 32 + half * 8;
      a.u[0] = packbf(pe[0],  pe[1]);
      a.u[1] = packbf(pe[2],  pe[3]);
      a.u[2] = packbf(pe[4],  pe[5]);
      a.u[3] = packbf(pe[6],  pe[7]);
      a.u[4] = packbf(pe[16], pe[17]);
      a.u[5] = packbf(pe[18], pe[19]);
      a.u[6] = packbf(pe[20], pe[21]);
      a.u[7] = packbf(pe[22], pe[23]);
      const uint4* pb = (const uint4*)(ktp + jt * 16 + half * 4);
      uint4 b0 = pb[0], b1 = pb[2];
      b.u[0] = b0.x; b.u[1] = b0.y; b.u[2] = b0.z; b.u[3] = b0.w;
      b.u[4] = b1.x; b.u[5] = b1.y; b.u[6] = b1.z; b.u[7] = b1.w;
      c = __builtin_amdgcn_wmma_f32_16x16x32_bf16(false, a.v, false, b.v, (short)0, c, false, false);
    }
    if (jpar == 1) {
#pragma unroll
      for (int v = 0; v < 8; ++v) dscr[(w4 * 32 + lane) * 8 + v] = c[v];
    }
    __syncthreads();
    if (jpar == 0) {
#pragma unroll
      for (int v = 0; v < 8; ++v) {
        float r = c[v] + dscr[(w4 * 32 + lane) * 8 + v];
        int n   = i0 + v + half * 8;
        int col = hd * HEAD_DIM + w4 * 16 + mrow;
        hprime[(size_t)n * OUT_DIM + col] = r + bias[col];
      }
    }
  }
}

// -------------------------------------------------------------------------
extern "C" void kernel_launch(void* const* d_in, const int* in_sizes, int n_in,
                              void* d_out, int out_size, void* d_ws, size_t ws_size,
                              hipStream_t stream) {
  const float* h    = (const float*)d_in[0];
  const float* adj  = (const float*)d_in[1];
  const float* Wsrc = (const float*)d_in[2];
  const float* Wdst = (const float*)d_in[3];
  const float* avec = (const float*)d_in[4];
  const float* bias = (const float*)d_in[5];

  float* hprime = (float*)d_out;
  float* attn   = (float*)d_out + (size_t)N_NODES * OUT_DIM;

  char* ws = (char*)d_ws;
  unsigned short* hb  = (unsigned short*)ws; ws += (size_t)N_NODES * IN_DIM * 2;
  unsigned short* WsT = (unsigned short*)ws; ws += (size_t)HEADS * IN_DIM * HEAD_DIM * 2;
  unsigned short* WdT = (unsigned short*)ws; ws += (size_t)HEADS * IN_DIM * HEAD_DIM * 2;
  float* Qf = (float*)ws;                    ws += (size_t)HEADS * N_NODES * HEAD_DIM * 4;
  float* Kf = (float*)ws;                    ws += (size_t)HEADS * N_NODES * HEAD_DIM * 4;
  unsigned short* KT = (unsigned short*)ws;  ws += (size_t)HEADS * HEAD_DIM * N_NODES * 2;

  (void)hipMemsetAsync(attn, 0, (size_t)N_NODES * N_NODES * sizeof(float), stream);

  k_cvt_h<<<(N_NODES * IN_DIM + 255) / 256, 256, 0, stream>>>(h, hb);
  k_cvt_w<<<(HEADS * IN_DIM * HEAD_DIM + 255) / 256, 256, 0, stream>>>(Wsrc, Wdst, WsT, WdT);
  k_proj<<<dim3(N_NODES / 16, HEADS), 128, 0, stream>>>(hb, WsT, WdT, Qf, Kf, KT);

  size_t lds_bytes = (size_t)LDS_FLOATS * sizeof(float);
  k_attn<<<dim3(N_NODES / 16, HEADS), 256, lds_bytes, stream>>>(
      Qf, Kf, KT, adj, avec, bias, hprime, attn);
}
